// Encoder_5368709120503
// MI455X (gfx1250) — matrix operand order repaired and checked by reading, not compiled, not run on hardware
//
#include <hip/hip_runtime.h>

// Problem constants (from reference)
#define D_FEAT   256
#define TWO_D    512
#define OUT_DIM  256
#define KNEI     16
#define BLOCK_M  64
// LDS tile strides chosen so row stride = 1040 B (=> 4 dwords mod 64 banks: conflict-free)
#define CS_STRIDE 520   // bf16 elems per combined row
#define HS_STRIDE 260   // f32 elems per h row

typedef __attribute__((ext_vector_type(16))) __bf16 v16bf;
typedef __attribute__((ext_vector_type(8)))  float  v8f;

union Frag { v16bf v; uint4 q[2]; };

__device__ __forceinline__ unsigned short f2bf(float x) {
    unsigned int u = __float_as_uint(x);
    u += 0x7FFFu + ((u >> 16) & 1u);          // round-to-nearest-even
    return (unsigned short)(u >> 16);
}

__global__ void w_to_bf16_kernel(const float* __restrict__ W,
                                 unsigned short* __restrict__ Wbf, int n) {
    int i = blockIdx.x * blockDim.x + threadIdx.x;
    if (i < n) Wbf[i] = f2bf(W[i]);
}

__global__ __launch_bounds__(256)
void sage_encoder_kernel(const float* __restrict__ feat,
                         const unsigned short* __restrict__ Wbf,   // [OUT][2D] bf16
                         const float* __restrict__ gamma,
                         const float* __restrict__ beta,
                         const int*   __restrict__ nodes,
                         const int*   __restrict__ neigh,          // [B][16]
                         float*       __restrict__ out,            // [OUT][B]
                         int Btot) {
    __shared__ __align__(16) unsigned char smem[BLOCK_M * CS_STRIDE * 2]; // 66560 B
    unsigned short* cs = (unsigned short*)smem;  // combined bf16 [64][520]
    float*          hs = (float*)smem;           // h f32 [64][260] (reuses same LDS)

    const int tid      = threadIdx.x;
    const int lane     = tid & 31;
    const int wave     = tid >> 5;
    const int l16      = lane & 15;
    const int lhalf    = lane >> 4;              // 0: lanes 0-15, 1: lanes 16-31
    const int block_b0 = blockIdx.x * BLOCK_M;

    // ---------------- Phase A: gather + mean -> bf16 combined tile ----------------
    for (int task = tid; task < BLOCK_M * (D_FEAT / 4); task += 256) {
        const int m = task >> 6;                 // row within block
        const int c = task & 63;                 // float4 chunk within D
        const int b = block_b0 + m;
        const int* nidx = neigh + (long)b * KNEI;

        float4 acc = make_float4(0.f, 0.f, 0.f, 0.f);
        #pragma unroll
        for (int j = 0; j < KNEI; ++j) {
            const float4 f = *(const float4*)(feat + (long)nidx[j] * D_FEAT + c * 4);
            acc.x += f.x; acc.y += f.y; acc.z += f.z; acc.w += f.w;
        }
        const float inv16 = 1.0f / 16.0f;
        const float4 s = *(const float4*)(feat + (long)nodes[b] * D_FEAT + c * 4);

        // neighbor mean -> cols [0,256), self -> cols [256,512)
        unsigned int p0 = (unsigned)f2bf(acc.x * inv16) | ((unsigned)f2bf(acc.y * inv16) << 16);
        unsigned int p1 = (unsigned)f2bf(acc.z * inv16) | ((unsigned)f2bf(acc.w * inv16) << 16);
        *(uint2*)(cs + m * CS_STRIDE + c * 4) = make_uint2(p0, p1);

        unsigned int q0 = (unsigned)f2bf(s.x) | ((unsigned)f2bf(s.y) << 16);
        unsigned int q1 = (unsigned)f2bf(s.z) | ((unsigned)f2bf(s.w) << 16);
        *(uint2*)(cs + m * CS_STRIDE + D_FEAT + c * 4) = make_uint2(q0, q1);
    }
    __syncthreads();

    // ---------------- Phase B: WMMA GEMM  h[64,256] = combined @ W^T ----------------
    const int wave_m = wave >> 2;                // 0..1  -> 32 rows each
    const int wave_n = wave & 3;                 // 0..3  -> 64 cols each

    v8f acc[2][4] = {};

    for (int k0 = 0; k0 < TWO_D; k0 += 32) {
        // A fragments (this code's contiguous K order, same as B): lanes 0-15 hold K k0..k0+15,
        // lanes 16-31 hold K k0+16..k0+31, row M = l16.
        Frag a[2];
        #pragma unroll
        for (int mi = 0; mi < 2; ++mi) {
            const unsigned short* rowp =
                cs + (wave_m * 32 + mi * 16 + l16) * CS_STRIDE + k0 + 16 * lhalf;
            a[mi].q[0] = *(const uint4*)(rowp);        // K k0..k0+7 (+16 if hi half)
            a[mi].q[1] = *(const uint4*)(rowp + 8);    // K k0+8..k0+15 (+16 if hi half)
        }
        // B fragments: lane = column (W row), contiguous 16 bf16 of K; hi half lanes K+16.
        Frag bf[4];
        #pragma unroll
        for (int ni = 0; ni < 4; ++ni) {
            const unsigned short* wp =
                Wbf + (size_t)(wave_n * 64 + ni * 16 + l16) * TWO_D + k0 + 16 * lhalf;
            bf[ni].v = *(const v16bf*)wp;
        }
        #pragma unroll
        for (int mi = 0; mi < 2; ++mi)
            #pragma unroll
            for (int ni = 0; ni < 4; ++ni)
                acc[mi][ni] = __builtin_amdgcn_wmma_f32_16x16x32_bf16(
                    false, a[mi].v, false, bf[ni].v, (short)0, acc[mi][ni], false, false);
    }

    __syncthreads();   // done reading combined tile; reuse LDS as f32 h tile

    // C/D layout: lanes 0-15 -> N=lane, M=vgpr r; lanes 16-31 -> N=lane-16, M=8+r.
    #pragma unroll
    for (int mi = 0; mi < 2; ++mi) {
        #pragma unroll
        for (int ni = 0; ni < 4; ++ni) {
            const int mb = wave_m * 32 + mi * 16 + lhalf * 8;
            const int nb = wave_n * 64 + ni * 16 + l16;
            #pragma unroll
            for (int r = 0; r < 8; ++r)
                hs[(mb + r) * HS_STRIDE + nb] = acc[mi][ni][r];
        }
    }
    __syncthreads();

    // ---------------- LayerNorm (unbiased std, eps on std) + ReLU, per row ----------------
    {
        const int m0 = wave * 8;                 // each wave owns 8 rows
        #pragma unroll 1
        for (int rr = 0; rr < 8; ++rr) {
            const int m = m0 + rr;
            float v[8], s = 0.f, s2 = 0.f;
            #pragma unroll
            for (int i = 0; i < 8; ++i) {
                v[i] = hs[m * HS_STRIDE + lane + 32 * i];
                s += v[i]; s2 += v[i] * v[i];
            }
            #pragma unroll
            for (int off = 16; off > 0; off >>= 1) {
                s  += __shfl_xor(s,  off, 32);
                s2 += __shfl_xor(s2, off, 32);
            }
            const float mean = s * (1.0f / 256.0f);
            const float var  = fmaxf((s2 - 256.0f * mean * mean) * (1.0f / 255.0f), 0.0f);
            const float inv  = 1.0f / (__builtin_sqrtf(var) + 1e-6f);
            #pragma unroll
            for (int i = 0; i < 8; ++i) {
                const int o = lane + 32 * i;
                float y = gamma[o] * (v[i] - mean) * inv + beta[o];
                hs[m * HS_STRIDE + o] = fmaxf(y, 0.0f);
            }
        }
    }
    __syncthreads();

    // ---------------- Transposed coalesced write-out: out[o][b] ----------------
    for (int task = tid; task < OUT_DIM * (BLOCK_M / 4); task += 256) {
        const int o  = task >> 4;
        const int b0 = (task & 15) * 4;
        float4 v;
        v.x = hs[(b0 + 0) * HS_STRIDE + o];
        v.y = hs[(b0 + 1) * HS_STRIDE + o];
        v.z = hs[(b0 + 2) * HS_STRIDE + o];
        v.w = hs[(b0 + 3) * HS_STRIDE + o];
        *(float4*)(out + (size_t)o * Btot + block_b0 + b0) = v;
    }
}

extern "C" void kernel_launch(void* const* d_in, const int* in_sizes, int n_in,
                              void* d_out, int out_size, void* d_ws, size_t ws_size,
                              hipStream_t stream) {
    const float* feat  = (const float*)d_in[0];   // [N_NODES, 256]
    const float* W     = (const float*)d_in[1];   // [256, 512]
    const float* gamma = (const float*)d_in[2];   // [256]
    const float* beta  = (const float*)d_in[3];   // [256]
    const int*   nodes = (const int*)d_in[4];     // [B]
    const int*   neigh = (const int*)d_in[5];     // [B, 16]
    float*       out   = (float*)d_out;           // [256, B]

    unsigned short* Wbf = (unsigned short*)d_ws;  // 256 KB bf16 copy of W
    const int wn   = in_sizes[1];                 // 131072
    const int Btot = in_sizes[4];                 // 65536

    w_to_bf16_kernel<<<(wn + 255) / 256, 256, 0, stream>>>(W, Wbf, wn);
    sage_encoder_kernel<<<Btot / BLOCK_M, 256, 0, stream>>>(
        feat, Wbf, gamma, beta, nodes, neigh, out, Btot);
}
